// MLPDecoder_35261681500175
// MI455X (gfx1250) — matrix operand order repaired and checked by reading, not compiled, not run on hardware
//
#include <hip/hip_runtime.h>

#define BB 4
#define BN 64
#define TT 32
#define DD 16
#define HH 128
#define KT 4
#define EE (BN * (BN - 1))

typedef __attribute__((ext_vector_type(16))) __bf16   v16bf;
typedef __attribute__((ext_vector_type(8)))  float    v8f;
typedef __attribute__((ext_vector_type(2)))  __bf16   bf16x2;
typedef __attribute__((ext_vector_type(4)))  unsigned uint4v;
typedef __attribute__((ext_vector_type(4)))  float    float4v;
typedef __attribute__((ext_vector_type(2)))  float    float2v;

union Frag { v16bf v; unsigned u[8]; uint4v q[2]; };

// native pack: lowers to v_cvt_pk_bf16_f32 (RNE) instead of manual bit math
__device__ inline unsigned pk_bf16(float lo, float hi) {
    bf16x2 p;
    p.x = (__bf16)lo;
    p.y = (__bf16)hi;
    return __builtin_bit_cast(unsigned, p);
}
__device__ inline unsigned short f2bf(float x) {
    __bf16 h = (__bf16)x;
    return __builtin_bit_cast(unsigned short, h);
}

__global__ __launch_bounds__(256)
void mlp_decoder_kernel(const float* __restrict__ x,     // (B,N,T,D)
                        const float* __restrict__ edges, // (B,E,K)
                        const float* __restrict__ W1,    // (K,2D,H)
                        const float* __restrict__ b1,    // (K,H)
                        const float* __restrict__ W2,    // (K,H,H)
                        const float* __restrict__ b2,    // (K,H)
                        const float* __restrict__ Wout,  // (H,D)
                        const float* __restrict__ bout,  // (D,)
                        float* __restrict__ out)         // (B,N,T-1,D)
{
    const int n = blockIdx.x;   // receiver node
    const int t = blockIdx.y;   // timestep 0..T-2
    const int b = blockIdx.z;   // batch

    __shared__ unsigned Ainp[64 * 16];      // 64 rows x 32 bf16 (A panel)
    __shared__ unsigned hbuf[2][16 * 64];   // double-buffered 16x128 bf16 h tile
    __shared__ float    ew[KT * 64];        // per-type per-row edge weights
    __shared__ float    aggr[HH];           // aggregated message for node n

    const int tid  = threadIdx.x;
    const int lane = tid & 31;
    const int wave = tid >> 5;              // 0..7 -> H-column tile j
    const int l15  = lane & 15;
    const int half = lane >> 4;

    // ---- stage edge-input panel: row r = [x[b,s_r,t,:] | x[b,n,t,:]], row 63 = 0 ----
    for (int idx = tid; idx < 64 * 16; idx += 256) {
        int r  = idx >> 4;       // edge row within node
        int cp = idx & 15;       // dword pair -> cols 2cp, 2cp+1 (never straddles D)
        float2v f = {0.f, 0.f};
        if (r < 63) {
            int s  = (r < n) ? r : r + 1;    // sender id
            int c0 = cp * 2;
            int a0 = (c0 < DD) ? (((b * BN + s) * TT + t) * DD + c0)
                               : (((b * BN + n) * TT + t) * DD + (c0 - DD));
            f = *(const float2v*)&x[a0];
        }
        Ainp[idx] = pk_bf16(f.x, f.y);
    }
    {   // edge weights, padded row -> 0
        int i = tid >> 6, r = tid & 63;
        float v = 0.f;
        if (r < 63) v = edges[(size_t)(b * EE + n * 63 + r) * KT + i];
        ew[tid] = v;
    }
    __syncthreads();

    const int col = wave * 16 + l15;        // this lane's hidden column (0..127)
    float aggrcol = 0.f;                    // running column partial

    for (int ty = 1; ty < KT; ++ty) {
        // ---- B1 fragment: W1[ty] (32 x 128). K = 2v + 16*half ----
        Frag B1;
        const float* W1p = W1 + (size_t)ty * (2 * DD) * HH;
#pragma unroll
        for (int v = 0; v < 8; ++v) {
            int k = 2 * v + 16 * half;
            B1.u[v] = pk_bf16(W1p[(size_t)k * HH + col], W1p[(size_t)(k + 1) * HH + col]);
        }
        // ---- B2 fragments: W2[ty] (128 x 128), 4 K-steps ----
        Frag B2[4];
        const float* W2p = W2 + (size_t)ty * HH * HH;
#pragma unroll
        for (int kk = 0; kk < 4; ++kk)
#pragma unroll
            for (int v = 0; v < 8; ++v) {
                int k = 32 * kk + 2 * v + 16 * half;
                B2[kk].u[v] = pk_bf16(W2p[(size_t)k * HH + col], W2p[(size_t)(k + 1) * HH + col]);
            }
        const float b1v = b1[ty * HH + col];
        const float b2v = b2[ty * HH + col];

        for (int tile = 0; tile < 4; ++tile) {
            const int p = tile & 1;
            // ---- A1 fragment: two contiguous 16B quads per lane ----
            Frag A1;
            {
                const unsigned* Arow = &Ainp[(tile * 16 + l15) * 16];
                A1.q[0] = *(const uint4v*)&Arow[8 * half];       // K 0..7  / 16..23
                A1.q[1] = *(const uint4v*)&Arow[8 * half + 4];   // K 8..15 / 24..31
            }
            // GEMM1: C1 = A1 @ W1-slab
            v8f c1 = {};
            c1 = __builtin_amdgcn_wmma_f32_16x16x32_bf16(false, A1.v, false, B1.v,
                                                         (short)0, c1, false, false);
            // ELU(bias) -> bf16 h tile (C layout: M = v + 8*half, N = col)
#pragma unroll
            for (int v = 0; v < 8; ++v) {
                float hv = c1[v] + b1v;
                hv = (hv > 0.f) ? hv : (__expf(hv) - 1.f);
                ((unsigned short*)hbuf[p])[(v + 8 * half) * HH + col] = f2bf(hv);
            }
            __syncthreads();
            // GEMM2: C2 = h(16x128) @ W2-slab, 4 chained WMMAs; A2 via b128 LDS loads
            v8f c2 = {};
#pragma unroll
            for (int kk = 0; kk < 4; ++kk) {
                Frag A2;
                const unsigned* Hrow = &hbuf[p][l15 * 64];
                A2.q[0] = *(const uint4v*)&Hrow[kk * 16 + 8 * half];
                A2.q[1] = *(const uint4v*)&Hrow[kk * 16 + 8 * half + 4];
                c2 = __builtin_amdgcn_wmma_f32_16x16x32_bf16(false, A2.v, false, B2[kk].v,
                                                             (short)0, c2, false, false);
            }
            // edge-weighted accumulate + folded row reduction (rows m = tile*16+v+8*half)
            float4v e0 = *(const float4v*)&ew[ty * 64 + tile * 16 + 8 * half];
            float4v e1 = *(const float4v*)&ew[ty * 64 + tile * 16 + 8 * half + 4];
#pragma unroll
            for (int v = 0; v < 4; ++v) {
                aggrcol += (c2[v]     + b2v) * e0[v];
                aggrcol += (c2[v + 4] + b2v) * e1[v];
            }
        }
    }

    // combine M0-7 (lanes 0-15) with M8-15 (lanes 16-31) -> full column sums
    aggrcol += __shfl_xor(aggrcol, 16, 32);
    if (half == 0) aggr[col] = aggrcol;
    __syncthreads();

    // final projection: out[b,n,t,d] = aggr @ Wout[:,d] + bout[d] + x[b,n,t,d]
    if (tid < DD) {
        float s = bout[tid];
#pragma unroll 8
        for (int hh = 0; hh < HH; ++hh) s += aggr[hh] * Wout[hh * DD + tid];
        s += x[((size_t)(b * BN + n) * TT + t) * DD + tid];
        out[((size_t)(b * BN + n) * (TT - 1) + t) * DD + tid] = s;
    }
}

extern "C" void kernel_launch(void* const* d_in, const int* in_sizes, int n_in,
                              void* d_out, int out_size, void* d_ws, size_t ws_size,
                              hipStream_t stream) {
    // setup_inputs order: x, edges, rel_rec, rel_send, W1, b1, W2, b2, Wout, bout
    const float* x     = (const float*)d_in[0];
    const float* edges = (const float*)d_in[1];
    const float* W1    = (const float*)d_in[4];
    const float* b1    = (const float*)d_in[5];
    const float* W2    = (const float*)d_in[6];
    const float* b2    = (const float*)d_in[7];
    const float* Wout  = (const float*)d_in[8];
    const float* bout  = (const float*)d_in[9];
    float* out = (float*)d_out;

    dim3 grid(BN, TT - 1, BB);   // (node, time, batch) = 64 x 31 x 4
    mlp_decoder_kernel<<<grid, 256, 0, stream>>>(x, edges, W1, b1, W2, b2,
                                                 Wout, bout, out);
}
